// Indexer_17600775979892
// MI455X (gfx1250) — matrix operand order repaired and checked by reading, not compiled, not run on hardware
//
#include <hip/hip_runtime.h>

// ---------------------------------------------------------------------------
// Problem constants (match reference)
// ---------------------------------------------------------------------------
#define BATCH   2
#define SEQ     2048
#define HID     2048
#define HEADS   16
#define HD      128
#define ROPE    64
#define MROWS   (BATCH * SEQ)          // 4096 total (b,s) rows
#define SCALE_F 0.022097086912079612f  // HEAD_DIM^-0.5 * HEADS^-0.5 = 2048^-0.5

typedef __attribute__((ext_vector_type(8)))  __bf16 v8bf;
typedef __attribute__((ext_vector_type(16))) __bf16 v16bf;
typedef __attribute__((ext_vector_type(8)))  float  v8f;

static __device__ __forceinline__ v16bf cat8(v8bf lo, v8bf hi) {
  return __builtin_shufflevector(lo, hi, 0,1,2,3,4,5,6,7,8,9,10,11,12,13,14,15);
}

static __device__ __forceinline__ v8f wmma_bf16(v16bf a, v16bf b, v8f c) {
  return __builtin_amdgcn_wmma_f32_16x16x32_bf16(
      false, a, false, b, (short)0, c, false, false);
}

// ---------------------------------------------------------------------------
// Kernel 1: fp32 -> bf16 conversion (grid-stride), for hidden_states (A side)
// ---------------------------------------------------------------------------
__global__ __launch_bounds__(256)
void cvt_f32_to_bf16(const float* __restrict__ in, __bf16* __restrict__ out, int n) {
  int i = blockIdx.x * blockDim.x + threadIdx.x;
  int stride = gridDim.x * blockDim.x;
  for (; i < n; i += stride) out[i] = (__bf16)in[i];
}

// ---------------------------------------------------------------------------
// Kernel 2: fused transpose + fp32 -> bf16 for weights.
// in: K x N (row-major fp32), out: N x K (row-major bf16).
// 16x16 LDS tile, coalesced on both sides. Grid = (K/16, N/16).
// ---------------------------------------------------------------------------
__global__ __launch_bounds__(256)
void transpose_f32_to_bf16(const float* __restrict__ in, __bf16* __restrict__ out,
                           int K, int N) {
  __shared__ float t[16][17];                 // pad to avoid bank conflicts
  const int k0 = blockIdx.x * 16;
  const int n0 = blockIdx.y * 16;
  const int tr = threadIdx.x >> 4;            // 0..15
  const int tc = threadIdx.x & 15;            // 0..15
  t[tr][tc] = in[(size_t)(k0 + tr) * N + n0 + tc];
  __syncthreads();
  out[(size_t)(n0 + tr) * K + k0 + tc] = (__bf16)t[tc][tr];
}

// ---------------------------------------------------------------------------
// Kernel 3: bf16 WMMA GEMM with pre-transposed B.
//   C(fp32, MxN) = A(MxK, row-major) * BT(NxK, row-major)^T
// Each wave computes MT x NT 16x16 tiles (8 independent accumulator chains
// for MT=2,NT=4). No LDS, no barriers: both A and B fragments are contiguous
// aligned per-lane global loads using one contiguous k order for A and B:
//   A 16x32 bf16: lane<16 -> row=lane,   K = 0..15 contiguous (+kc*32)
//                 lane>=16 -> row=lane-16, K = 16..31 contiguous
//   B 32x16 bf16: lane<16 -> col=lane,   K = 0..15 contiguous (+kc*32)
//                 lane>=16 -> col=lane-16, K = 16..31 contiguous
// Block = 8 waves stacked along M. Grid = (N/(NT*16), M/(8*MT*16)).
// ---------------------------------------------------------------------------
template<int MT, int NT>
__global__ __launch_bounds__(256)
void gemm_bf16_tn(const __bf16* __restrict__ A, const __bf16* __restrict__ BT,
                  float* __restrict__ C, int N, int K) {
  const int tid   = threadIdx.x;
  const int wave  = tid >> 5;
  const int lane  = tid & 31;
  const int row16 = lane & 15;
  const int hf    = lane >> 4;
  const int m0    = (blockIdx.y * 8 + wave) * (MT * 16);
  const int n0    = blockIdx.x * (NT * 16);

  const __bf16* aBase[MT];
  const __bf16* bBase[NT];
#pragma unroll
  for (int mt = 0; mt < MT; ++mt)
    aBase[mt] = A + (size_t)(m0 + mt * 16 + row16) * K + hf * 16;
#pragma unroll
  for (int nt = 0; nt < NT; ++nt)
    bBase[nt] = BT + (size_t)(n0 + nt * 16 + row16) * K + hf * 16;

  v8f acc[MT][NT] = {};
  const int kIters = K >> 5;
  for (int kc = 0; kc < kIters; ++kc) {
    v16bf a[MT], b[NT];
#pragma unroll
    for (int mt = 0; mt < MT; ++mt) {
      const __bf16* ap = aBase[mt] + kc * 32;
      a[mt] = *(const v16bf*)ap;
    }
#pragma unroll
    for (int nt = 0; nt < NT; ++nt)
      b[nt] = *(const v16bf*)(bBase[nt] + kc * 32);

    __builtin_prefetch(aBase[0] + kc * 32 + 256, 0, 3);  // ~8 K-steps ahead
    __builtin_prefetch(bBase[0] + kc * 32 + 256, 0, 3);

#pragma unroll
    for (int mt = 0; mt < MT; ++mt)
#pragma unroll
      for (int nt = 0; nt < NT; ++nt)
        acc[mt][nt] = wmma_bf16(a[mt], b[nt], acc[mt][nt]);
  }

  // D layout: VGPR r -> M = r + 8*hf, N = row16
#pragma unroll
  for (int mt = 0; mt < MT; ++mt) {
    float* cp = C + (size_t)(m0 + mt * 16 + hf * 8) * N + n0 + row16;
#pragma unroll
    for (int nt = 0; nt < NT; ++nt)
#pragma unroll
      for (int r = 0; r < 8; ++r)
        cp[(size_t)r * N + nt * 16] = acc[mt][nt][r];
  }
}

// ---------------------------------------------------------------------------
// Kernel 4: epilogue — LayerNorm(k) + RoPE(q, k) + scale(w); emits bf16 q/k.
// One block (128 threads) per (b,s) row; thread d owns head-dim d.
// ---------------------------------------------------------------------------
__global__ __launch_bounds__(128)
void epilogue_kernel(const float* __restrict__ qraw, const float* __restrict__ kraw,
                     const float* __restrict__ wraw, const float* __restrict__ cosb,
                     const float* __restrict__ sinb, const float* __restrict__ gamma,
                     const float* __restrict__ beta,
                     __bf16* __restrict__ qout, __bf16* __restrict__ kout,
                     float* __restrict__ wout) {
  __shared__ float red[128];
  __shared__ float kn[128];
  const int row = blockIdx.x;        // b*SEQ + s
  const int d   = threadIdx.x;       // 0..127

  // ---- LayerNorm over k row ----
  const float kv = kraw[(size_t)row * HD + d];
  red[d] = kv; __syncthreads();
  for (int off = 64; off; off >>= 1) {
    if (d < off) red[d] += red[d + off];
    __syncthreads();
  }
  const float mu = red[0] * (1.0f / HD);
  __syncthreads();
  const float dv = kv - mu;
  red[d] = dv * dv; __syncthreads();
  for (int off = 64; off; off >>= 1) {
    if (d < off) red[d] += red[d + off];
    __syncthreads();
  }
  const float var  = red[0] * (1.0f / HD);
  const float rstd = rsqrtf(var + 1e-5f);
  const float nrm  = dv * rstd * gamma[d] + beta[d];
  kn[d] = nrm; __syncthreads();

  float cd = 0.f, sd = 0.f;
  if (d < ROPE) {
    cd = cosb[(size_t)row * ROPE + d];
    sd = sinb[(size_t)row * ROPE + d];
  }

  // ---- RoPE on k ----
  float kvo = nrm;
  if (d < ROPE) {
    const float part = (d < 32) ? -kn[d + 32] : kn[d - 32];
    kvo = nrm * cd + part * sd;
  }
  kout[(size_t)row * HD + d] = (__bf16)kvo;

  // ---- RoPE on q (all heads, dim d) ----
  const float* qr = qraw + (size_t)row * (HEADS * HD);
  __bf16*      qo = qout + (size_t)row * (HEADS * HD);
#pragma unroll
  for (int h = 0; h < HEADS; ++h) {
    const float qv = qr[h * HD + d];
    float r = qv;
    if (d < ROPE) {
      const float part = (d < 32) ? -qr[h * HD + d + 32] : qr[h * HD + d - 32];
      r = qv * cd + part * sd;
    }
    qo[h * HD + d] = (__bf16)r;
  }

  // ---- scale w ----
  if (d < HEADS) {
    wout[(size_t)row * HEADS + d] = SCALE_F * wraw[(size_t)row * HEADS + d];
  }
}

// ---------------------------------------------------------------------------
// Kernel 5: fused score.  out[b,q,k] = sum_h w[b,q,h] * relu(q_h . k)
// grid = (SEQ/256, SEQ/16, BATCH); block = 256 (8 waves).
// Each wave owns a 16q x 32k strip (two 16x16 tiles).  B fragments for both
// k-tiles are hoisted into registers across the whole head loop; per head the
// A fragment is loaded once per K-chunk and feeds two independent WMMA
// chains (s0, s1).  relu + w-weighted head reduction happens in registers via
// v_readlane broadcast of a per-lane-preloaded w fragment.  No LDS, no
// barriers: all 8 waves of a block share the same 16-row q tile through WGP$.
// ---------------------------------------------------------------------------
__global__ __launch_bounds__(256)
void score_kernel(const __bf16* __restrict__ Q, const __bf16* __restrict__ Km,
                  const float* __restrict__ W, float* __restrict__ Out) {
  const int tid   = threadIdx.x;
  const int wave  = tid >> 5;
  const int lane  = tid & 31;
  const int row16 = lane & 15;
  const int hf    = lane >> 4;
  const int b     = blockIdx.z;
  const int q0    = blockIdx.y * 16;
  const int k0    = (blockIdx.x * 8 + wave) * 32;
  const size_t qrow = (size_t)b * SEQ + q0;
  const size_t krow = (size_t)b * SEQ + k0;

  // ---- hoisted B fragments for both k-tiles (k rows are contiguous K) ----
  v16bf bfr[2][4];
#pragma unroll
  for (int j = 0; j < 2; ++j) {
    const __bf16* kp = Km + (krow + j * 16 + row16) * HD + hf * 16;
#pragma unroll
    for (int kc = 0; kc < 4; ++kc)
      bfr[j][kc] = *(const v16bf*)(kp + kc * 32);   // 32B aligned
  }

  // ---- per-lane w fragment: wv[r] in lane l holds w[q0 + r + 8*(l>=16)][l&15]
  //      so readlane(wv[r], h) / readlane(wv[r], 16+h) broadcast w[.,h] ----
  float wv[8];
  const float* wp = W + (qrow + hf * 8) * HEADS + row16;
#pragma unroll
  for (int r = 0; r < 8; ++r) wv[r] = wp[r * HEADS];

  v8f acc0 = {}, acc1 = {};
  const __bf16* aB = Q + (qrow + row16) * (size_t)(HEADS * HD) + hf * 16;

#pragma unroll
  for (int h = 0; h < HEADS; ++h) {
    v8f s0 = {}, s1 = {};
#pragma unroll
    for (int kc = 0; kc < 4; ++kc) {
      const __bf16* ap = aB + h * HD + kc * 32;
      v16bf afr = *(const v16bf*)ap;
      s0 = wmma_bf16(afr, bfr[0][kc], s0);    // two independent chains
      s1 = wmma_bf16(afr, bfr[1][kc], s1);
    }
#pragma unroll
    for (int r = 0; r < 8; ++r) {
      const int ilo = __builtin_amdgcn_readlane(__builtin_bit_cast(int, wv[r]), h);
      const int ihi = __builtin_amdgcn_readlane(__builtin_bit_cast(int, wv[r]), 16 + h);
      const float wsel = __builtin_bit_cast(float, hf ? ihi : ilo);
      float sv0 = s0[r]; sv0 = sv0 > 0.f ? sv0 : 0.f;
      float sv1 = s1[r]; sv1 = sv1 > 0.f ? sv1 : 0.f;
      acc0[r] += wsel * sv0;
      acc1[r] += wsel * sv1;
    }
  }

  // ---- store: row M = q0 + r + 8*hf, cols k0 + row16 and k0 + 16 + row16 ----
  float* op = Out + ((size_t)b * SEQ + q0 + hf * 8) * SEQ + k0 + row16;
#pragma unroll
  for (int r = 0; r < 8; ++r) {
    op[(size_t)r * SEQ]      = acc0[r];
    op[(size_t)r * SEQ + 16] = acc1[r];
  }
}

// ---------------------------------------------------------------------------
// Host launcher
// ---------------------------------------------------------------------------
extern "C" void kernel_launch(void* const* d_in, const int* in_sizes, int n_in,
                              void* d_out, int out_size, void* d_ws, size_t ws_size,
                              hipStream_t stream) {
  (void)in_sizes; (void)n_in; (void)out_size; (void)ws_size;

  const float* hs    = (const float*)d_in[0];
  const float* cosb  = (const float*)d_in[1];
  const float* sinb  = (const float*)d_in[2];
  const float* Wq    = (const float*)d_in[3];
  const float* Wk    = (const float*)d_in[4];
  const float* Ww    = (const float*)d_in[5];
  const float* gamma = (const float*)d_in[6];
  const float* beta  = (const float*)d_in[7];
  float* out = (float*)d_out;

  // ---- carve workspace (≈80 MB) ----
  char* base = (char*)d_ws;
  size_t off = 0;
  auto carve = [&](size_t bytes) -> char* {
    char* r = base + off;
    off += (bytes + 255) & ~(size_t)255;
    return r;
  };
  __bf16* hs_bf = (__bf16*)carve((size_t)MROWS * HID * 2);
  __bf16* wqT   = (__bf16*)carve((size_t)(HEADS * HD) * HID * 2);  // 2048 x 2048
  __bf16* wkT   = (__bf16*)carve((size_t)HD * HID * 2);            // 128 x 2048
  __bf16* wwT   = (__bf16*)carve((size_t)HEADS * HID * 2);         // 16 x 2048
  float*  q_raw = (float*)carve((size_t)MROWS * (HEADS * HD) * 4);
  float*  k_raw = (float*)carve((size_t)MROWS * HD * 4);
  float*  w_raw = (float*)carve((size_t)MROWS * HEADS * 4);
  __bf16* q_bf  = (__bf16*)carve((size_t)MROWS * (HEADS * HD) * 2);
  __bf16* k_bf  = (__bf16*)carve((size_t)MROWS * HD * 2);
  float*  w_scl = (float*)carve((size_t)MROWS * HEADS * 4);

  // ---- 1) A-side fp32 -> bf16 ----
  cvt_f32_to_bf16<<<2048, 256, 0, stream>>>(hs, hs_bf, MROWS * HID);

  // ---- 2) weight transpose + convert (B-operand order) ----
  transpose_f32_to_bf16<<<dim3(HID / 16, (HEADS * HD) / 16), 256, 0, stream>>>(
      Wq, wqT, HID, HEADS * HD);
  transpose_f32_to_bf16<<<dim3(HID / 16, HD / 16), 256, 0, stream>>>(
      Wk, wkT, HID, HD);
  transpose_f32_to_bf16<<<dim3(HID / 16, HEADS / 16), 256, 0, stream>>>(
      Ww, wwT, HID, HEADS);

  // ---- 3) projections via WMMA GEMM (M = 4096, K = 2048) ----
  gemm_bf16_tn<2, 4><<<dim3((HEADS * HD) / 64, MROWS / 256), 256, 0, stream>>>(
      hs_bf, wqT, q_raw, HEADS * HD, HID);
  gemm_bf16_tn<2, 4><<<dim3(HD / 64, MROWS / 256), 256, 0, stream>>>(
      hs_bf, wkT, k_raw, HD, HID);
  gemm_bf16_tn<2, 1><<<dim3(HEADS / 16, MROWS / 256), 256, 0, stream>>>(
      hs_bf, wwT, w_raw, HEADS, HID);

  // ---- 4) LayerNorm + RoPE + scale epilogue ----
  epilogue_kernel<<<MROWS, 128, 0, stream>>>(
      q_raw, k_raw, w_raw, cosb, sinb, gamma, beta, q_bf, k_bf, w_scl);

  // ---- 5) fused relu/weighted score ----
  score_kernel<<<dim3(SEQ / 256, SEQ / 16, BATCH), 256, 0, stream>>>(
      q_bf, k_bf, w_scl, out);
}
